// LSTMBlock_2190433321425
// MI455X (gfx1250) — matrix operand order repaired and checked by reading, not compiled, not run on hardware
//
#include <hip/hip_runtime.h>
#include <hip/hip_bf16.h>

// ---------------------------------------------------------------------------
// ConvLSTM stack for MI455X (gfx1250): every conv lowered to implicit GEMM on
// v_wmma_f32_16x16x32_bf16 (bf16 in, f32 accumulate). Wave32, one wave per
// 16(M)x64(N) output tile of the z = [i|f|c|o] gate pre-activations.
// Geometry is templated so the im2col K-decode constant-folds away.
// ---------------------------------------------------------------------------

typedef __bf16 bf16;
typedef __attribute__((ext_vector_type(16))) __bf16 v16bf;
typedef __attribute__((ext_vector_type(8)))  float  v8f;

struct __align__(16) U4 { unsigned int a, b, c, d; };
union Frag32 { v16bf v; U4 q[2]; };

__device__ __forceinline__ float hsig(float v) {
    return fminf(fmaxf(0.2f * v + 0.5f, 0.0f), 1.0f);
}

// ---------------------------------------------------------------------------
// prep: transpose + cast conv weights (kh,kw,Cin,256) f32 -> [256][Kpad] bf16
// (zero padded so K is a multiple of 32)
// ---------------------------------------------------------------------------
__global__ void prep_w(const float* __restrict__ src, bf16* __restrict__ dst,
                       int Ktot, int Kpad) {
    int idx = blockIdx.x * blockDim.x + threadIdx.x;
    if (idx >= 256 * Kpad) return;
    int n = idx / Kpad;
    int k = idx - n * Kpad;
    dst[idx] = (k < Ktot) ? (bf16)src[(long long)k * 256 + n] : (bf16)0.0f;
}

__global__ void cast_f32_bf16(const float* __restrict__ src,
                              bf16* __restrict__ dst, int n) {
    int idx = blockIdx.x * blockDim.x + threadIdx.x;
    if (idx < n) dst[idx] = (bf16)src[idx];
}

// ---------------------------------------------------------------------------
// Implicit-GEMM conv via WMMA, fully templated geometry.
//   z[M=B*Ho*Wo][256] (f32)  +=/=  im2col(src)[M][K] * Wt[K][256]
// One wave: M-tile = 16 pixels (blockIdx.x), N-tile = 64 cols (blockIdx.y*64),
// 4 accumulators reusing one A fragment per 32-wide K step.
// bias != nullptr : acc initialized with bias (input-conv pass)
// bias == nullptr : acc initialized from z (recurrent accumulate pass)
// ---------------------------------------------------------------------------
template <int HoWo, int Wo, int Hin, int Win, int Cin, int KW,
          int STRIDE, int PAD, int Ktot, int Kpad>
__global__ __launch_bounds__(32) void convlstm_gemm(
    const bf16* __restrict__ src,   // feature map, NHWC, bf16
    const bf16* __restrict__ wK,    // [256][Kpad] bf16 (pre-transposed)
    const float* __restrict__ bias, // [256] or nullptr
    float* __restrict__ z,          // [M][256] f32
    long long bStride, long long tOff)  // src elem offsets: base = b*bStride+tOff
{
    const int lane = threadIdx.x;        // 0..31
    const int hf   = lane >> 4;          // half-wave select
    const int ln   = lane & 15;

    const int mt    = blockIdx.x;
    const int nbase = blockIdx.y * 64;

    // A-fragment pixel for this lane (ISA 7.12.2: A row m = lane%16)
    const int m   = mt * 16 + ln;
    const int b   = m / HoWo;            // HoWo is a power of two -> shift
    const int rem = m - b * HoWo;
    const int oy  = rem / Wo;            // Wo is a power of two -> shift
    const int ox  = rem - oy * Wo;
    const int iy0 = oy * STRIDE - PAD;
    const int ix0 = ox * STRIDE - PAD;
    const long long pbase = (long long)b * bStride + tOff;

    // C/D rows for this lane: m = mt*16 + hf*8 + r ; col = nbase + j*16 + ln
    const long long zrow = ((long long)mt * 16 + hf * 8) * 256 + nbase + ln;

    v8f acc[4];
    if (bias) {
#pragma unroll
        for (int j = 0; j < 4; j++) {
            float bv = bias[nbase + j * 16 + ln];
#pragma unroll
            for (int r = 0; r < 8; r++) acc[j][r] = bv;
        }
    } else {
#pragma unroll
        for (int j = 0; j < 4; j++)
#pragma unroll
            for (int r = 0; r < 8; r++)
                acc[j][r] = z[zrow + j * 16 + (long long)r * 256];
    }

    // B rows: lane reads K chunks [hf*8, +8) and [16+hf*8, +8) of column n
    const bf16* wrow0 = wK + (long long)(nbase + ln) * Kpad + hf * 8;
#pragma unroll
    for (int j = 0; j < 4; j++)
        __builtin_prefetch(wrow0 + (long long)j * 16 * Kpad, 0, 1);

    const U4 z4 = {0u, 0u, 0u, 0u};

#pragma unroll
    for (int kb = 0; kb < Kpad; kb += 32) {
        // ---- A fragment: K chunks [kb+hf*8, +8) and [kb+16+hf*8, +8) ----
        Frag32 a;
#pragma unroll
        for (int h8 = 0; h8 < 2; h8++) {
            const int k = kb + h8 * 16;          // compile-time constant
            U4 val = z4;
            if (k < Ktot) {
                const int kk = k / Cin;          // tap id (constant)
                const int ky = kk / KW;          // constants after unroll
                const int kx = kk - ky * KW;
                const int c  = k - kk * Cin;     // 8-aligned channel base
                int iy = iy0 + ky;
                int ix = ix0 + kx;
                if ((unsigned)iy < (unsigned)Hin && (unsigned)ix < (unsigned)Win) {
                    // per-lane extra K offset hf*8 stays inside one tap since
                    // Cin is a multiple of 16
                    val = *(const U4*)(src + pbase +
                                       (long long)(iy * Win + ix) * Cin + c + hf * 8);
                }
            }
            a.q[h8] = val;
        }

        // ---- 4 N-tiles share the A fragment ----
#pragma unroll
        for (int j = 0; j < 4; j++) {
            const bf16* wr = wrow0 + (long long)j * 16 * Kpad;
            Frag32 bb;
            bb.q[0] = *(const U4*)(wr + kb);
            bb.q[1] = *(const U4*)(wr + kb + 16);
            acc[j] = __builtin_amdgcn_wmma_f32_16x16x32_bf16(
                false, a.v, false, bb.v, (short)0, acc[j], false, false);
        }
    }

#pragma unroll
    for (int j = 0; j < 4; j++)
#pragma unroll
        for (int r = 0; r < 8; r++)
            z[zrow + j * 16 + (long long)r * 256] = acc[j][r];
}

// ---------------------------------------------------------------------------
// Gate update + optional folded BatchNorm, writes bf16 h into the sequence.
//   i = hs(zi); f = hs(zf); c = f*c_prev + i*relu(zc); h = hs(zo)*relu(c)
// ---------------------------------------------------------------------------
__global__ void gate_kernel(const float* __restrict__ zb,
                            float* __restrict__ cst,
                            bf16* __restrict__ hout,
                            const float* __restrict__ g,
                            const float* __restrict__ be,
                            int Npix, int HW,
                            long long bStride, long long tOff, int firstT) {
    int idx = blockIdx.x * blockDim.x + threadIdx.x;
    if (idx >= Npix * 64) return;
    int p = idx >> 6;
    int f = idx & 63;

    const float* zp = zb + (long long)p * 256;
    float zi = zp[f];
    float zf = zp[64 + f];
    float zc = zp[128 + f];
    float zo = zp[192 + f];

    float cp = firstT ? 0.0f : cst[idx];
    float ig = hsig(zi);
    float fg = hsig(zf);
    float cn = fg * cp + ig * fmaxf(zc, 0.0f);
    float h  = hsig(zo) * fmaxf(cn, 0.0f);
    cst[idx] = cn;

    float o = h;
    if (g) o = h * (g[f] * rsqrtf(1.0f + 1e-3f)) + be[f];

    int b  = p / HW;
    int yx = p - b * HW;
    hout[b * bStride + tOff + (long long)yx * 64 + f] = (bf16)o;
}

// ---------------------------------------------------------------------------
// Final Conv3D (3x3x3, 64 -> 1, SAME) + ReLU. Tiny: 113 MMACs on VALU.
// ---------------------------------------------------------------------------
__global__ void conv3d_relu(const bf16* __restrict__ in,   // (4,16,32,32,64)
                            const float* __restrict__ w,   // (3,3,3,64,1)
                            const float* __restrict__ bias,
                            float* __restrict__ out) {     // (4,16,32,32)
    int idx = blockIdx.x * blockDim.x + threadIdx.x;
    if (idx >= 4 * 16 * 32 * 32) return;
    int x = idx & 31;
    int y = (idx >> 5) & 31;
    int t = (idx >> 10) & 15;
    int b = idx >> 14;

    float acc = bias[0];
    for (int dt = 0; dt < 3; dt++) {
        int tt = t + dt - 1;
        if ((unsigned)tt >= 16u) continue;
        for (int dy = 0; dy < 3; dy++) {
            int yy = y + dy - 1;
            if ((unsigned)yy >= 32u) continue;
            for (int dx = 0; dx < 3; dx++) {
                int xx = x + dx - 1;
                if ((unsigned)xx >= 32u) continue;
                const bf16* ip = in + ((((long long)b * 16 + tt) * 32 + yy) * 32 + xx) * 64;
                const float* wp = w + ((dt * 3 + dy) * 3 + dx) * 64;
#pragma unroll 8
                for (int ci = 0; ci < 64; ci++)
                    acc += (float)ip[ci] * wp[ci];
            }
        }
    }
    out[idx] = fmaxf(acc, 0.0f);
}

// ---------------------------------------------------------------------------
extern "C" void kernel_launch(void* const* d_in, const int* in_sizes, int n_in,
                              void* d_out, int out_size, void* d_ws, size_t ws_size,
                              hipStream_t stream) {
    (void)in_sizes; (void)n_in; (void)out_size; (void)ws_size;

    const float* x    = (const float*)d_in[0];
    const float* k1   = (const float*)d_in[1];
    const float* rk1  = (const float*)d_in[2];
    const float* b1   = (const float*)d_in[3];
    const float* g1   = (const float*)d_in[4];
    const float* be1  = (const float*)d_in[5];
    const float* k2   = (const float*)d_in[6];
    const float* rk2  = (const float*)d_in[7];
    const float* b2   = (const float*)d_in[8];
    const float* g2   = (const float*)d_in[9];
    const float* be2  = (const float*)d_in[10];
    const float* k3   = (const float*)d_in[11];
    const float* rk3  = (const float*)d_in[12];
    const float* b3   = (const float*)d_in[13];
    const float* w3d  = (const float*)d_in[14];
    const float* b3db = (const float*)d_in[15];

    char* ws = (char*)d_ws;
    size_t off = 0;
    auto take = [&](size_t bytes) -> char* {
        char* p = ws + off;
        off += (bytes + 255) & ~(size_t)255;
        return p;
    };

    bf16* wbf1  = (bf16*)take((size_t)256 * 416 * 2);    // K=400 -> pad 416
    bf16* wrbf1 = (bf16*)take((size_t)256 * 1600 * 2);   // K=1600
    bf16* wbf2  = (bf16*)take((size_t)256 * 576 * 2);    // K=576
    bf16* wrbf2 = (bf16*)take((size_t)256 * 576 * 2);
    bf16* wbf3  = (bf16*)take((size_t)256 * 64 * 2);     // K=64
    bf16* wrbf3 = (bf16*)take((size_t)256 * 64 * 2);
    bf16* xbf   = (bf16*)take((size_t)4 * 16 * 64 * 64 * 16 * 2);
    bf16* seqA  = (bf16*)take((size_t)4 * 16 * 64 * 64 * 64 * 2);
    bf16* seqB  = (bf16*)take((size_t)4 * 16 * 64 * 64 * 64 * 2);
    float* cst  = (float*)take((size_t)4 * 64 * 64 * 64 * 4);
    float* zbuf = (float*)take((size_t)4 * 64 * 64 * 256 * 4);
    bf16* seqC  = (bf16*)seqA;   // layer-1 sequence is dead once layer 2 ran

    // --- weight prep + input cast ---
    cast_f32_bf16<<<(4194304 + 255) / 256, 256, 0, stream>>>(x, xbf, 4194304);
    prep_w<<<(256 * 416  + 255) / 256, 256, 0, stream>>>(k1,  wbf1,  400,  416);
    prep_w<<<(256 * 1600 + 255) / 256, 256, 0, stream>>>(rk1, wrbf1, 1600, 1600);
    prep_w<<<(256 * 576  + 255) / 256, 256, 0, stream>>>(k2,  wbf2,  576,  576);
    prep_w<<<(256 * 576  + 255) / 256, 256, 0, stream>>>(rk2, wrbf2, 576,  576);
    prep_w<<<(256 * 64   + 255) / 256, 256, 0, stream>>>(k3,  wbf3,  64,   64);
    prep_w<<<(256 * 64   + 255) / 256, 256, 0, stream>>>(rk3, wrbf3, 64,   64);

    dim3 blk(32);

    // --- Layer 1: ConvLSTM 5x5, Cin=16, 64x64, stride 1 ---
    {
        const int Npix = 4 * 64 * 64;
        dim3 grid(Npix / 16, 4);
        const long long bIn = (long long)16 * 64 * 64 * 16;
        const long long bH  = (long long)16 * 64 * 64 * 64;
        for (int t = 0; t < 16; t++) {
            convlstm_gemm<4096, 64, 64, 64, 16, 5, 1, 2, 400, 416>
                <<<grid, blk, 0, stream>>>(
                    xbf, wbf1, b1, zbuf, bIn, (long long)t * 64 * 64 * 16);
            if (t > 0)
                convlstm_gemm<4096, 64, 64, 64, 64, 5, 1, 2, 1600, 1600>
                    <<<grid, blk, 0, stream>>>(
                        seqA, wrbf1, nullptr, zbuf, bH,
                        (long long)(t - 1) * 64 * 64 * 64);
            gate_kernel<<<(Npix * 64 + 255) / 256, 256, 0, stream>>>(
                zbuf, cst, seqA, g1, be1, Npix, 64 * 64, bH,
                (long long)t * 64 * 64 * 64, t == 0);
        }
    }

    // --- Layer 2: ConvLSTM 3x3, Cin=64, 64x64, stride 1 ---
    {
        const int Npix = 4 * 64 * 64;
        dim3 grid(Npix / 16, 4);
        const long long bH = (long long)16 * 64 * 64 * 64;
        for (int t = 0; t < 16; t++) {
            convlstm_gemm<4096, 64, 64, 64, 64, 3, 1, 1, 576, 576>
                <<<grid, blk, 0, stream>>>(
                    seqA, wbf2, b2, zbuf, bH, (long long)t * 64 * 64 * 64);
            if (t > 0)
                convlstm_gemm<4096, 64, 64, 64, 64, 3, 1, 1, 576, 576>
                    <<<grid, blk, 0, stream>>>(
                        seqB, wrbf2, nullptr, zbuf, bH,
                        (long long)(t - 1) * 64 * 64 * 64);
            gate_kernel<<<(Npix * 64 + 255) / 256, 256, 0, stream>>>(
                zbuf, cst, seqB, g2, be2, Npix, 64 * 64, bH,
                (long long)t * 64 * 64 * 64, t == 0);
        }
    }

    // --- Layer 3: ConvLSTM 1x1, Cin=64, stride 2 -> 32x32 ---
    {
        const int Npix = 4 * 32 * 32;
        dim3 grid(Npix / 16, 4);
        const long long bIn = (long long)16 * 64 * 64 * 64;  // seqB per-b stride
        const long long bH  = (long long)16 * 32 * 32 * 64;  // seqC per-b stride
        for (int t = 0; t < 16; t++) {
            convlstm_gemm<1024, 32, 64, 64, 64, 1, 2, 0, 64, 64>
                <<<grid, blk, 0, stream>>>(
                    seqB, wbf3, b3, zbuf, bIn, (long long)t * 64 * 64 * 64);
            if (t > 0)
                convlstm_gemm<1024, 32, 32, 32, 64, 1, 1, 0, 64, 64>
                    <<<grid, blk, 0, stream>>>(
                        seqC, wrbf3, nullptr, zbuf, bH,
                        (long long)(t - 1) * 32 * 32 * 64);
            gate_kernel<<<(Npix * 64 + 255) / 256, 256, 0, stream>>>(
                zbuf, cst, seqC, nullptr, nullptr, Npix, 32 * 32, bH,
                (long long)t * 32 * 32 * 64, t == 0);
        }
    }

    // --- Conv3D + ReLU -> d_out (4,16,32,32,1) f32 ---
    conv3d_relu<<<(65536 + 255) / 256, 256, 0, stream>>>(
        seqC, w3d, b3db, (float*)d_out);
}
